// FactorizedEmbeddingsOutput_22273700397184
// MI455X (gfx1250) — compile-verified, matrix operand order checked by reading, not run on hardware
//
#include <hip/hip_runtime.h>

// Problem sizes (fixed by the reference)
#define B_DIM 4096
#define E_DIM 1024
#define M_DIM 4096
#define V_DIM 32000
#define K_DIM 8

typedef __attribute__((ext_vector_type(16))) __bf16  v16bf;
typedef __attribute__((ext_vector_type(8)))  float   v8f;
typedef __attribute__((ext_vector_type(4)))  unsigned int u32x4;

union Frag {
    v16bf bf;
    u32x4 u[2];
};

// ---------------------------------------------------------------------------
// fp32 -> (bf16 hi, bf16 lo) split, round-to-nearest-even
// ---------------------------------------------------------------------------
__device__ __forceinline__ unsigned short f32_to_bf16_rne(float f) {
    unsigned int u = __float_as_uint(f);
    unsigned int r = (u + 0x7FFFu + ((u >> 16) & 1u)) >> 16;
    return (unsigned short)r;
}

__global__ void convert_split_kernel(const float* __restrict__ src,
                                     unsigned short* __restrict__ hi,
                                     unsigned short* __restrict__ lo,
                                     int n) {
    int i = blockIdx.x * 256 + threadIdx.x;
    if (i < n) {
        float f = src[i];
        unsigned short h = f32_to_bf16_rne(f);
        hi[i] = h;
        float fh = __uint_as_float(((unsigned int)h) << 16);
        lo[i] = f32_to_bf16_rne(f - fh);
    }
}

// ---------------------------------------------------------------------------
// scores = x @ W^T via bf16x3 WMMA
//  - block: 256 threads = 8 waves, block tile 128(M) x 128(N)
//  - wave tile: 32(M) x 64(N) = 2x4 fragments of 16x16, K-step = 32
// A-fragment per-lane layout (16-bit A, 16x32):
//   lane<16 : row M = lane,    K in [k0..k0+7] and [k0+16..k0+23]
//   lane>=16: row M = lane-16, K in [k0+8..k0+15] and [k0+24..k0+31]
// B uses the symmetric layout read from W rows (W is N x K row-major).
// ---------------------------------------------------------------------------
__device__ __forceinline__ void load_frag(Frag& f, const unsigned short* __restrict__ base,
                                          int row, int kcol) {
    const unsigned short* p = base + (size_t)row * E_DIM + kcol;
    f.u[0] = *(const u32x4*)p;          // 8 bf16 = 16 bytes
    f.u[1] = *(const u32x4*)(p + 16);   // next 8 bf16, +16 elements
}

__global__ void __launch_bounds__(256)
gemm_bf16x3_kernel(const unsigned short* __restrict__ Ahi,
                   const unsigned short* __restrict__ Alo,
                   const unsigned short* __restrict__ Bhi,
                   const unsigned short* __restrict__ Blo,
                   float* __restrict__ C) {
    const int lane = threadIdx.x & 31;
    const int wave = threadIdx.x >> 5;
    const int wm   = wave & 3;      // 0..3  -> 4 * 32 = 128 rows
    const int wn   = wave >> 2;     // 0..1  -> 2 * 64 = 128 cols

    const int m0 = blockIdx.y * 128 + wm * 32;
    const int n0 = blockIdx.x * 128 + wn * 64;

    const int lr    = lane & 15;        // row within 16x16 tile
    const int half  = lane >> 4;        // 0 or 1
    const int kbase = half ? 8 : 0;     // per-lane K start inside K-step

    v8f acc[2][4];
#pragma unroll
    for (int i = 0; i < 2; ++i)
#pragma unroll
        for (int j = 0; j < 4; ++j)
            acc[i][j] = (v8f){0.f, 0.f, 0.f, 0.f, 0.f, 0.f, 0.f, 0.f};

    for (int kt = 0; kt < E_DIM; kt += 32) {
        Frag ah[2], al[2], bh[4], bl[4];
        const int kc = kt + kbase;

#pragma unroll
        for (int i = 0; i < 2; ++i) {
            const int row = m0 + i * 16 + lr;
            load_frag(ah[i], Ahi, row, kc);
            load_frag(al[i], Alo, row, kc);
        }
#pragma unroll
        for (int j = 0; j < 4; ++j) {
            const int row = n0 + j * 16 + lr;   // W row = GEMM column
            load_frag(bh[j], Bhi, row, kc);
            load_frag(bl[j], Blo, row, kc);
        }

#pragma unroll
        for (int i = 0; i < 2; ++i) {
#pragma unroll
            for (int j = 0; j < 4; ++j) {
                // small cross terms first, dominant term last
                acc[i][j] = __builtin_amdgcn_wmma_f32_16x16x32_bf16(
                    false, al[i].bf, false, bh[j].bf, (short)0, acc[i][j], false, false);
                acc[i][j] = __builtin_amdgcn_wmma_f32_16x16x32_bf16(
                    false, ah[i].bf, false, bl[j].bf, (short)0, acc[i][j], false, false);
                acc[i][j] = __builtin_amdgcn_wmma_f32_16x16x32_bf16(
                    false, ah[i].bf, false, bh[j].bf, (short)0, acc[i][j], false, false);
            }
        }
    }

    // C/D layout: VGPR r, lanes 0-15 -> (M = r,   N = lane)
    //                      lanes16-31 -> (M = r+8, N = lane-16)
#pragma unroll
    for (int i = 0; i < 2; ++i) {
#pragma unroll
        for (int j = 0; j < 4; ++j) {
            const int mbase = m0 + i * 16 + 8 * half;
            const int nbase = n0 + j * 16 + lr;
#pragma unroll
            for (int r = 0; r < 8; ++r) {
                C[(size_t)(mbase + r) * M_DIM + nbase] = acc[i][j][r];
            }
        }
    }
}

// ---------------------------------------------------------------------------
// out[b, v] = sum_k scores[b, idx[v, k]]
//  - each block: 256 v's x 8 b's; idx staged (transposed) in LDS
//  - scores rows (16KB) are L2-resident; output writes fully coalesced
// ---------------------------------------------------------------------------
__global__ void __launch_bounds__(256)
gather_sum_kernel(const float* __restrict__ scores,
                  const long long* __restrict__ idx,
                  float* __restrict__ out) {
    __shared__ int sidx[K_DIM * 256];   // sidx[k*256 + v_local]

    const int v0 = blockIdx.x * 256;
    const int b0 = blockIdx.y * 8;
    const int t  = threadIdx.x;

    // cooperative load of idx[v0 .. v0+255][0..7], stored transposed
    for (int i = t; i < 256 * K_DIM; i += 256) {
        const int vloc = i >> 3;        // i / 8
        const int k    = i & 7;         // i % 8
        sidx[k * 256 + vloc] = (int)idx[(size_t)(v0 + vloc) * K_DIM + k];
    }
    __syncthreads();

    const int v = v0 + t;
#pragma unroll
    for (int bb = 0; bb < 8; ++bb) {
        const float* srow = scores + (size_t)(b0 + bb) * M_DIM;
        float s = 0.f;
#pragma unroll
        for (int k = 0; k < K_DIM; ++k) {
            s += srow[sidx[k * 256 + t]];
        }
        out[(size_t)(b0 + bb) * V_DIM + v] = s;
    }
}

// ---------------------------------------------------------------------------
// launch
// ---------------------------------------------------------------------------
extern "C" void kernel_launch(void* const* d_in, const int* in_sizes, int n_in,
                              void* d_out, int out_size, void* d_ws, size_t ws_size,
                              hipStream_t stream) {
    const float*     x   = (const float*)d_in[0];      // (B, E) fp32
    const float*     W   = (const float*)d_in[1];      // (M, E) fp32
    const long long* idx = (const long long*)d_in[2];  // (V, K) int64
    float* out = (float*)d_out;                        // (B, V) fp32

    char* ws = (char*)d_ws;
    const size_t SCORES_BYTES = (size_t)B_DIM * M_DIM * sizeof(float);   // 64 MB
    const size_t HALF_BYTES   = (size_t)B_DIM * E_DIM * sizeof(unsigned short); // 8 MB each

    float*          scores = (float*)ws;
    unsigned short* xhi = (unsigned short*)(ws + SCORES_BYTES);
    unsigned short* xlo = (unsigned short*)(ws + SCORES_BYTES + HALF_BYTES);
    unsigned short* whi = (unsigned short*)(ws + SCORES_BYTES + 2 * HALF_BYTES);
    unsigned short* wlo = (unsigned short*)(ws + SCORES_BYTES + 3 * HALF_BYTES);

    const int nElem = B_DIM * E_DIM;   // 4,194,304 (same for x and W)
    const int cvtBlocks = (nElem + 255) / 256;

    convert_split_kernel<<<cvtBlocks, 256, 0, stream>>>(x, xhi, xlo, nElem);
    convert_split_kernel<<<cvtBlocks, 256, 0, stream>>>(W, whi, wlo, nElem);

    dim3 gemmGrid(M_DIM / 128, B_DIM / 128);   // (32, 32)
    gemm_bf16x3_kernel<<<gemmGrid, 256, 0, stream>>>(xhi, xlo, whi, wlo, scores);

    dim3 gatherGrid(V_DIM / 256, B_DIM / 8);   // (125, 512)
    gather_sum_kernel<<<gatherGrid, 256, 0, stream>>>(scores, idx, out);
}
